// ConvBlock_35570919145962
// MI455X (gfx1250) — 8-bit matrix operand order repaired and checked by reading, compiled, not run on hardware
//
#include <hip/hip_runtime.h>


typedef int v8i __attribute__((ext_vector_type(8)));

#define NBATCH 32
#define CIN    128
#define HH     56
#define WW     56
#define COUT   256
#define POS    (HH*WW)
#define ROWS   (NBATCH*POS)
#define KTOT   (CIN*9)
#define KSTEPS 18

#define QX_OFF 0u
#define QX_SZ  ((size_t)ROWS*CIN)
#define QW_OFF (QX_OFF + QX_SZ)
#define QW_SZ  ((size_t)KSTEPS*16*1024)
#define S_OFF  (QW_OFF + QW_SZ)
#define S_SZ   ((size_t)ROWS*4)
#define WM_OFF (S_OFF + S_SZ)

__global__ void quant_x_kernel(const float* __restrict__ x,
                               unsigned char* __restrict__ qx) {
  __shared__ unsigned char t[32][33];
  const int n       = blockIdx.z;
  const int posBase = blockIdx.x * 32;
  const int ciBase  = blockIdx.y * 32;
  const int tx = threadIdx.x, ty = threadIdx.y;
#pragma unroll
  for (int i = 0; i < 4; ++i) {
    const int ci = ciBase + ty + i * 8;
    float v = x[((size_t)(n * CIN + ci)) * POS + posBase + tx];
    v = fminf(fmaxf(v, 0.f), 4.f) * 63.75f;
    t[ty + i * 8][tx] = (unsigned char)(int)rintf(v);
  }
  __syncthreads();
#pragma unroll
  for (int i = 0; i < 4; ++i) {
    const int p = posBase + ty + i * 8;
    qx[((size_t)n * POS + p) * CIN + ciBase + tx] = t[tx][ty + i * 8];
  }
}

__global__ void wmax_init(unsigned* wm) { *wm = 0u; }

__global__ void wmax_kernel(const float* __restrict__ w, unsigned* wm, int nElem) {
  float m = 0.f;
  for (int i = blockIdx.x * blockDim.x + threadIdx.x; i < nElem;
       i += gridDim.x * blockDim.x)
    m = fmaxf(m, fabsf(tanhf(w[i])));
  __shared__ float red[256];
  red[threadIdx.x] = m;
  __syncthreads();
  for (int s = 128; s > 0; s >>= 1) {
    if (threadIdx.x < s) red[threadIdx.x] = fmaxf(red[threadIdx.x], red[threadIdx.x + s]);
    __syncthreads();
  }
  if (threadIdx.x == 0) atomicMax(wm, __float_as_uint(red[0]));
}

__global__ void pack_w_kernel(const float* __restrict__ w,
                              const unsigned* __restrict__ wm,
                              char* __restrict__ qwp) {
  const int i = blockIdx.x * blockDim.x + threadIdx.x;
  if (i >= COUT * KTOT) return;
  const float m = __uint_as_float(*wm);
  const int co = i / KTOT;
  const int r  = i % KTOT;
  const int ci = r / 9;
  const int tp = r % 9;
  const float wt = tanhf(w[i]) / m;
  int q = (int)rintf((wt + 1.f) * 127.5f);
  q = min(max(q, 0), 255);
  const int k  = tp * CIN + ci;
  const int kb = k >> 6, kr = k & 63;
  const int nb = co >> 4, nl = co & 15;
  const int g = kr >> 4, rr = kr & 15;
  const int lane = nl + (g & 1) * 16;
  const int j = (g >> 1) * 4 + (rr >> 2);
  const int b = rr & 3;
  qwp[(((size_t)kb * 16 + nb) * 32 + lane) * 32 + j * 4 + b] = (char)(q - 128);
}

__device__ inline unsigned byteSum(unsigned v, unsigned acc) {
#if __has_builtin(__builtin_amdgcn_udot4)
  return __builtin_amdgcn_udot4(v, 0x01010101u, acc, false);
#else
  return acc + (v & 0xff) + ((v >> 8) & 0xff) + ((v >> 16) & 0xff) + (v >> 24);
#endif
}

__global__ void rowsum_kernel(const unsigned char* __restrict__ qx,
                              int* __restrict__ S) {
  const int r = blockIdx.x * blockDim.x + threadIdx.x;
  if (r >= ROWS) return;
  const int n = r / POS, p = r % POS, y = p / WW, x = p % WW;
  unsigned s = 0;
#pragma unroll
  for (int t = 0; t < 9; ++t) {
    const int ys = y + t / 3 - 1, xs = x + t % 3 - 1;
    if ((unsigned)ys < HH && (unsigned)xs < WW) {
      const uint4* q =
          (const uint4*)(qx + ((size_t)(n * POS + ys * WW + xs) << 7));
#pragma unroll
      for (int d = 0; d < 8; ++d) {
        const uint4 v = q[d];
        s = byteSum(v.x, s); s = byteSum(v.y, s);
        s = byteSum(v.z, s); s = byteSum(v.w, s);
      }
    }
  }
  S[r] = (int)s;
}

__device__ inline v8i loadB(const char* p) {
  const int4 u = *(const int4*)p;
  const int4 v = *(const int4*)(p + 16);
  v8i b;
  b[0] = u.x; b[1] = u.y; b[2] = u.z; b[3] = u.w;
  b[4] = v.x; b[5] = v.y; b[6] = v.z; b[7] = v.w;
  return b;
}

__global__ __launch_bounds__(256)
void conv_wmma_kernel(const unsigned char* __restrict__ qx,
                      const char* __restrict__ qwp,
                      const int* __restrict__ S,
                      const float* __restrict__ bias,
                      const float* __restrict__ gamma,
                      const float* __restrict__ beta,
                      const float* __restrict__ rmean,
                      const float* __restrict__ rvar,
                      float* __restrict__ out) {
  const int tid  = threadIdx.x;
  const int wave = tid >> 5, l = tid & 31, m = l & 15, hi = l >> 4;
  const int rowBase = blockIdx.x * 128 + wave * 16;
  const int chBase  = blockIdx.y * 64;

  const int r = rowBase + m;
  const int n = r / POS, p = r % POS, y = p / WW, x = p % WW;

  v8i acc0 = {}, acc1 = {}, acc2 = {}, acc3 = {};
  const char* pBbase = qwp + ((size_t)(chBase >> 4)) * 1024 + (size_t)l * 32;

  for (int kb = 0; kb < KSTEPS; ++kb) {
    const int tap = kb >> 1;
    const int ys = y + tap / 3 - 1, xs = x + tap % 3 - 1;
    const bool valid = ((unsigned)ys < HH) && ((unsigned)xs < WW);
    const char* pA = (const char*)qx +
                     (((size_t)(n * POS + ys * WW + xs)) << 7) +
                     (kb & 1) * 64 + hi * 16;
    const int2 z  = make_int2(0, 0);
    const int2 c0 = valid ? *(const int2*)(pA +  0) : z;
    const int2 c1 = valid ? *(const int2*)(pA +  8) : z;
    const int2 c2 = valid ? *(const int2*)(pA + 32) : z;
    const int2 c3 = valid ? *(const int2*)(pA + 40) : z;
    v8i a;
    a[0] = c0.x; a[1] = c0.y; a[2] = c1.x; a[3] = c1.y;
    a[4] = c2.x; a[5] = c2.y; a[6] = c3.x; a[7] = c3.y;

    const char* pB = pBbase + (size_t)kb * 16 * 1024;
    const v8i b0 = loadB(pB);
    const v8i b1 = loadB(pB + 1024);
    const v8i b2 = loadB(pB + 2048);
    const v8i b3 = loadB(pB + 3072);

    acc0 = __builtin_amdgcn_wmma_i32_16x16x64_iu8(false, a, true, b0, acc0, false, false);
    acc1 = __builtin_amdgcn_wmma_i32_16x16x64_iu8(false, a, true, b1, acc1, false, false);
    acc2 = __builtin_amdgcn_wmma_i32_16x16x64_iu8(false, a, true, b2, acc2, false, false);
    acc3 = __builtin_amdgcn_wmma_i32_16x16x64_iu8(false, a, true, b3, acc3, false, false);
  }

  const float scale = 4.0f / 65025.0f;
  float sPart[8];
  size_t outOff[8];
#pragma unroll
  for (int v = 0; v < 8; ++v) {
    const int rr = rowBase + v + hi * 8;
    sPart[v] = (float)S[rr];
    const int nn = rr / POS, pp = rr % POS;
    outOff[v] = (size_t)nn * COUT * POS + pp;
  }
#pragma unroll
  for (int f = 0; f < 4; ++f) {
    const int co = chBase + f * 16 + m;
    const float inv = gamma[co] * rsqrtf(rvar[co] + 1e-5f);
    const float shf = beta[co] - rmean[co] * inv + bias[co];
    const v8i acc = (f == 0) ? acc0 : (f == 1) ? acc1 : (f == 2) ? acc2 : acc3;
#pragma unroll
    for (int v = 0; v < 8; ++v) {
      const float val = (2.0f * (float)acc[v] + sPart[v]) * scale * inv + shf;
      out[outOff[v] + (size_t)co * POS] = val;
    }
  }
}

extern "C" void kernel_launch(void* const* d_in, const int* in_sizes, int n_in,
                              void* d_out, int out_size, void* d_ws, size_t ws_size,
                              hipStream_t stream) {
  (void)in_sizes; (void)n_in; (void)out_size; (void)ws_size;
  const float* x     = (const float*)d_in[0];
  const float* w     = (const float*)d_in[1];
  const float* bias  = (const float*)d_in[2];
  const float* gamma = (const float*)d_in[3];
  const float* beta  = (const float*)d_in[4];
  const float* rmean = (const float*)d_in[5];
  const float* rvar  = (const float*)d_in[6];

  char* ws = (char*)d_ws;
  unsigned char* qx  = (unsigned char*)(ws + QX_OFF);
  char*          qwp = ws + QW_OFF;
  int*           S   = (int*)(ws + S_OFF);
  unsigned*      wm  = (unsigned*)(ws + WM_OFF);

  dim3 b1(32, 8), g1(POS / 32, CIN / 32, NBATCH);
  quant_x_kernel<<<g1, b1, 0, stream>>>(x, qx);

  wmax_init<<<1, 1, 0, stream>>>(wm);
  wmax_kernel<<<288, 256, 0, stream>>>(w, wm, COUT * KTOT);
  pack_w_kernel<<<(COUT * KTOT + 255) / 256, 256, 0, stream>>>(w, wm, qwp);

  rowsum_kernel<<<(ROWS + 255) / 256, 256, 0, stream>>>(qx, S);

  dim3 g6(ROWS / 128, COUT / 64);
  conv_wmma_kernel<<<g6, 256, 0, stream>>>(qx, qwp, S, bias, gamma, beta,
                                           rmean, rvar, (float*)d_out);
}
